// MultiHeadAttention_27779848470561
// MI455X (gfx1250) — matrix operand order repaired and checked by reading, not compiled, not run on hardware
//
#include <hip/hip_runtime.h>
#include <hip/hip_bf16.h>

#define DMODEL 1024
#define NHEAD  16
#define DK     64
#define BATCH  4
#define SEQ    2048
#define SR     2068   // padded LDS row stride (floats): row*20 mod 64 spreads banks

typedef __attribute__((ext_vector_type(16))) __bf16 v16bf;
typedef __attribute__((ext_vector_type(8)))  __bf16 v8bf;
typedef __attribute__((ext_vector_type(8)))  float  v8f;
typedef __attribute__((ext_vector_type(4)))  float  v4f;

// ---------------------------------------------------------------------------
// Weight conversion f32 -> bf16
// ---------------------------------------------------------------------------
__global__ void convert_w_kernel(const float* __restrict__ in,
                                 __bf16* __restrict__ out, int n) {
  int i = blockIdx.x * blockDim.x + threadIdx.x;
  if (i < n) out[i] = (__bf16)in[i];
}

// ---------------------------------------------------------------------------
// Projection GEMM: Y[m,n] = sum_k X[m,k] * W[n,k]   (x @ W^T)
//   mode 0: out bf16, head layout    [B,H,S,DK]   (for Q, K)
//   mode 1: out bf16, transposed     [B,H,DK,S]   (for V)
//   mode 2: out f32,  flat [M,1024] + bias        (for Wo)
// Block = 128 threads = 4 waves; each wave computes a 16x64 strip.
// ---------------------------------------------------------------------------
__global__ __launch_bounds__(128)
void proj_kernel(const float* __restrict__ X, const __bf16* __restrict__ Wb,
                 void* __restrict__ out, const float* __restrict__ bias,
                 int mode) {
  const int lane = threadIdx.x & 31;
  const int wave = threadIdx.x >> 5;
  const int half = lane >> 4;
  const int l16  = lane & 15;

  const int m0 = blockIdx.x * 64 + wave * 16;
  const int n0 = blockIdx.y * 64;

  v8f acc[4];
  const v8f vzero = {0.f,0.f,0.f,0.f,0.f,0.f,0.f,0.f};
#pragma unroll
  for (int i = 0; i < 4; ++i) acc[i] = vzero;

  const float* Xrow = X + (size_t)(m0 + l16) * DMODEL;

  for (int k = 0; k < DMODEL; k += 32) {
    const int kb = k + half * 16;
    v4f a0 = *reinterpret_cast<const v4f*>(Xrow + kb);
    v4f a1 = *reinterpret_cast<const v4f*>(Xrow + kb + 4);
    v4f a2 = *reinterpret_cast<const v4f*>(Xrow + kb + 8);
    v4f a3 = *reinterpret_cast<const v4f*>(Xrow + kb + 12);
    v16bf a;
#pragma unroll
    for (int i = 0; i < 4; ++i) {
      a[i]      = (__bf16)a0[i];
      a[4 + i]  = (__bf16)a1[i];
      a[8 + i]  = (__bf16)a2[i];
      a[12 + i] = (__bf16)a3[i];
    }
    const int krow = k + half * 16;
#pragma unroll
    for (int nt = 0; nt < 4; ++nt) {
      const int n = n0 + nt * 16 + l16;
      v16bf bfrag = *reinterpret_cast<const v16bf*>(Wb + (size_t)n * DMODEL + krow);
      acc[nt] = __builtin_amdgcn_wmma_f32_16x16x32_bf16(
          false, a, false, bfrag, (short)0, acc[nt], false, false);
    }
  }

#pragma unroll
  for (int nt = 0; nt < 4; ++nt) {
#pragma unroll
    for (int r = 0; r < 8; ++r) {
      const int m = m0 + half * 8 + r;
      const int n = n0 + nt * 16 + l16;
      const float v = acc[nt][r];
      if (mode == 2) {
        ((float*)out)[(size_t)m * DMODEL + n] = v + bias[n];
      } else {
        const int b = m >> 11, s = m & (SEQ - 1);
        const int h = n >> 6,  d = n & (DK - 1);
        const size_t idx = (mode == 0)
            ? ((size_t)((b * NHEAD + h) * SEQ + s)) * DK + d
            : ((size_t)((b * NHEAD + h) * DK + d)) * SEQ + s;
        ((__bf16*)out)[idx] = (__bf16)v;
      }
    }
  }
}

// ---------------------------------------------------------------------------
// Fused attention. One workgroup = (batch b, 16 query rows), looping over all
// 16 heads: the mask tile is staged ONCE to LDS (transposed bytes) and reused
// 16x (mask HBM traffic 1.07 GB -> 67 MB). Score panel rows padded to SR=2068
// floats for bank-conflict-free softmax / WMMA fragment access.
// Attention output leaves via global_store_async_from_lds_b128 (ASYNCcnt DMA)
// overlapped with the context WMMA GEMM.
// ---------------------------------------------------------------------------
__global__ __launch_bounds__(256)
void attn_kernel(const __bf16* __restrict__ Qh, const __bf16* __restrict__ Kh,
                 const __bf16* __restrict__ Vt, const int* __restrict__ mask,
                 float* __restrict__ attn_out, float* __restrict__ ctx) {
  extern __shared__ char smem_raw[];
  float* sc     = (float*)smem_raw;   // 16 rows x SR floats (padded panel)
  float* red    = sc + 16 * SR;       // 16*16 reduction scratch
  float* rowinv = red + 256;          // 16 reciprocal sums
  float* ctile  = rowinv + 16;        // 4*256 split-K reduction
  unsigned char* mbT = (unsigned char*)(ctile + 1024);  // [2048][16] mask bytes

  const int lane = threadIdx.x & 31;
  const int wave = threadIdx.x >> 5;
  const int half = lane >> 4;
  const int l16  = lane & 15;

  const int q0 = blockIdx.x * 16;
  const int b  = blockIdx.y;

  const v8f vzero = {0.f,0.f,0.f,0.f,0.f,0.f,0.f,0.f};

  // ---- Phase 0: stage mask tile, transposed, as bytes ----
  const int* mrow = mask + ((size_t)b * SEQ + q0) * SEQ;
  for (int idx = threadIdx.x; idx < 16 * SEQ; idx += 256) {
    const int m = idx >> 11, n = idx & (SEQ - 1);
    mbT[n * 16 + m] = (unsigned char)(mrow[idx] != 0);
  }
  __syncthreads();

  const unsigned sc_lds_base = (unsigned)(size_t)sc;  // LDS byte address

  for (int h = 0; h < NHEAD; ++h) {
    const __bf16* Qb = Qh + ((size_t)(b * NHEAD + h) * SEQ) * DK;
    const __bf16* Kb = Kh + ((size_t)(b * NHEAD + h) * SEQ) * DK;
    const __bf16* Vb = Vt + ((size_t)(b * NHEAD + h) * DK) * SEQ;

    // ---- Phase 1: scores -> LDS ----
    const __bf16* qrow = Qb + (size_t)(q0 + l16) * DK + half * 16;
    v8bf qa = *reinterpret_cast<const v8bf*>(qrow);
    v8bf qb = *reinterpret_cast<const v8bf*>(qrow + 8);
    v8bf qc = *reinterpret_cast<const v8bf*>(qrow + 32);
    v8bf qd = *reinterpret_cast<const v8bf*>(qrow + 40);
    v16bf aq0, aq1;
#pragma unroll
    for (int i = 0; i < 8; ++i) {
      aq0[i] = qa[i]; aq0[8 + i] = qb[i];
      aq1[i] = qc[i]; aq1[8 + i] = qd[i];
    }

#pragma unroll 2
    for (int nt = 0; nt < 16; ++nt) {
      const int n0 = wave * 256 + nt * 16;
      const int n  = n0 + l16;
      const __bf16* kp = Kb + (size_t)n * DK + half * 16;
      v16bf b0 = *reinterpret_cast<const v16bf*>(kp);
      v16bf b1 = *reinterpret_cast<const v16bf*>(kp + 32);
      v8f acc = vzero;
      acc = __builtin_amdgcn_wmma_f32_16x16x32_bf16(false, aq0, false, b0,
                                                    (short)0, acc, false, false);
      acc = __builtin_amdgcn_wmma_f32_16x16x32_bf16(false, aq1, false, b1,
                                                    (short)0, acc, false, false);
      // 8 mask bytes for this column (rows half*8 .. half*8+7) in one load.
      const unsigned long long mw =
          *reinterpret_cast<const unsigned long long*>(mbT + (size_t)n * 16 + half * 8);
#pragma unroll
      for (int r = 0; r < 8; ++r) {
        const int m = half * 8 + r;
        const unsigned mk = (unsigned)(mw >> (8 * r)) & 0xffu;
        sc[m * SR + n] = mk ? acc[r] * 0.125f : -1.0e9f;
      }
    }
    __syncthreads();

    // ---- Phase 2: softmax over each of the 16 rows ----
    const int row = threadIdx.x >> 4;
    const int sub = threadIdx.x & 15;

    float mx = -3.0e38f;
    for (int j = sub; j < SEQ; j += 16) mx = fmaxf(mx, sc[row * SR + j]);
    red[row * 16 + sub] = mx;
    __syncthreads();
    mx = red[row * 16];
#pragma unroll
    for (int t = 1; t < 16; ++t) mx = fmaxf(mx, red[row * 16 + t]);
    __syncthreads();

    float sum = 0.f;
    for (int j = sub; j < SEQ; j += 16) {
      const float e = __expf(sc[row * SR + j] - mx);
      sc[row * SR + j] = e;
      sum += e;
    }
    red[row * 16 + sub] = sum;
    __syncthreads();
    sum = 0.f;
#pragma unroll
    for (int t = 0; t < 16; ++t) sum += red[row * 16 + t];
    if (sub == 0) rowinv[row] = 1.0f / sum;
    __syncthreads();

    // Normalize panel in LDS (coalesced per row segment).
    for (int idx = threadIdx.x; idx < 16 * SEQ; idx += 256) {
      const int m = idx >> 11, n = idx & (SEQ - 1);
      sc[m * SR + n] = sc[m * SR + n] * rowinv[m];
    }
    __syncthreads();

    // ---- Issue async DMA: LDS panel -> global attention output ----
    // 16*2048 floats; 256 threads x 32 iters x 16 B per instruction.
    {
      float* aout = attn_out + ((size_t)(b * NHEAD + h) * SEQ + q0) * SEQ;
#pragma unroll 4
      for (int i = 0; i < 32; ++i) {
        const int f0  = (i * 256 + (int)threadIdx.x) * 4;   // global float idx
        const int m   = f0 >> 11, n = f0 & (SEQ - 1);
        const unsigned lds_addr = sc_lds_base + (unsigned)(m * SR + n) * 4u;
        const unsigned long long gaddr =
            (unsigned long long)(size_t)(aout + f0);
        asm volatile("global_store_async_from_lds_b128 %0, %1, off"
                     :: "v"(gaddr), "v"(lds_addr)
                     : "memory");
      }
    }

    // ---- Phase 3 (overlaps DMA): context = attn @ V, split-K over waves ----
    const int nt    = wave & 3;   // 16-wide d tile
    const int khalf = wave >> 2;  // K half (1024 each)
    v8f cacc = vzero;
    const __bf16* vcol = Vb + (size_t)(nt * 16 + l16) * SEQ;
    const int kend = khalf * 1024 + 1024;
#pragma unroll 2
    for (int kb = khalf * 1024; kb < kend; kb += 32) {
      const int kbase = kb + half * 16;
      const float* sp = &sc[(size_t)l16 * SR + kbase];
      v4f c0 = *reinterpret_cast<const v4f*>(sp);
      v4f c1 = *reinterpret_cast<const v4f*>(sp + 4);
      v4f c2 = *reinterpret_cast<const v4f*>(sp + 8);
      v4f c3 = *reinterpret_cast<const v4f*>(sp + 12);
      v16bf a;
#pragma unroll
      for (int i = 0; i < 4; ++i) {
        a[i]      = (__bf16)c0[i];
        a[4 + i]  = (__bf16)c1[i];
        a[8 + i]  = (__bf16)c2[i];
        a[12 + i] = (__bf16)c3[i];
      }
      v16bf bf = *reinterpret_cast<const v16bf*>(vcol + kb + half * 16);
      cacc = __builtin_amdgcn_wmma_f32_16x16x32_bf16(false, a, false, bf,
                                                     (short)0, cacc, false, false);
    }
    if (wave >= 4) {
#pragma unroll
      for (int r = 0; r < 8; ++r) ctile[nt * 256 + r * 32 + lane] = cacc[r];
    }
    __syncthreads();
    if (wave < 4) {
#pragma unroll
      for (int r = 0; r < 8; ++r) {
        const float v = cacc[r] + ctile[nt * 256 + r * 32 + lane];
        const int m = half * 8 + r;
        ctx[((size_t)b * SEQ + q0 + m) * DMODEL + h * DK + nt * 16 + l16] = v;
      }
    }

    // Drain this head's attention DMA before the panel is overwritten.
    asm volatile("s_wait_asynccnt 0x0" ::: "memory");
    __syncthreads();
  }
}

// ---------------------------------------------------------------------------
extern "C" void kernel_launch(void* const* d_in, const int* in_sizes, int n_in,
                              void* d_out, int out_size, void* d_ws, size_t ws_size,
                              hipStream_t stream) {
  const float* q    = (const float*)d_in[0];
  const float* k    = (const float*)d_in[1];
  const float* v    = (const float*)d_in[2];
  const int*   mask = (const int*)  d_in[3];
  const float* Wq   = (const float*)d_in[4];
  const float* Wk   = (const float*)d_in[5];
  const float* Wv   = (const float*)d_in[6];
  const float* Wo   = (const float*)d_in[7];
  const float* bo   = (const float*)d_in[8];

  char* ws = (char*)d_ws;
  __bf16* Wqb = (__bf16*)ws; ws += (size_t)DMODEL * DMODEL * 2;
  __bf16* Wkb = (__bf16*)ws; ws += (size_t)DMODEL * DMODEL * 2;
  __bf16* Wvb = (__bf16*)ws; ws += (size_t)DMODEL * DMODEL * 2;
  __bf16* Wob = (__bf16*)ws; ws += (size_t)DMODEL * DMODEL * 2;
  __bf16* Qh  = (__bf16*)ws; ws += (size_t)BATCH * SEQ * DMODEL * 2;
  __bf16* Kh  = (__bf16*)ws; ws += (size_t)BATCH * SEQ * DMODEL * 2;
  __bf16* Vt  = (__bf16*)ws; ws += (size_t)BATCH * SEQ * DMODEL * 2;
  float*  ctx = (float*) ws;

  float* out_ctx  = (float*)d_out;
  float* out_attn = out_ctx + (size_t)BATCH * SEQ * DMODEL;

  // 1. Weights to bf16.
  const int wn = DMODEL * DMODEL;
  convert_w_kernel<<<wn / 256, 256, 0, stream>>>(Wq, Wqb, wn);
  convert_w_kernel<<<wn / 256, 256, 0, stream>>>(Wk, Wkb, wn);
  convert_w_kernel<<<wn / 256, 256, 0, stream>>>(Wv, Wvb, wn);
  convert_w_kernel<<<wn / 256, 256, 0, stream>>>(Wo, Wob, wn);

  // 2. Q/K/V projections.
  dim3 pgrid(BATCH * SEQ / 64, DMODEL / 64);
  proj_kernel<<<pgrid, 128, 0, stream>>>(q, Wqb, Qh, nullptr, 0);
  proj_kernel<<<pgrid, 128, 0, stream>>>(k, Wkb, Kh, nullptr, 0);
  proj_kernel<<<pgrid, 128, 0, stream>>>(v, Wvb, Vt, nullptr, 1);

  // 3. Fused scores + softmax + context (loops over heads internally).
  size_t smem = (size_t)(16 * SR + 256 + 16 + 1024) * sizeof(float)
              + (size_t)16 * SEQ;  // + transposed mask bytes
  attn_kernel<<<dim3(SEQ / 16, BATCH), 256, smem, stream>>>(
      Qh, Kh, Vt, mask, out_attn, ctx);

  // 4. Output projection + bias.
  proj_kernel<<<pgrid, 128, 0, stream>>>(ctx, Wob, (void*)out_ctx, bo, 2);
}
